// MPQBertOutputp_90692529422554
// MI455X (gfx1250) — compile-verified, matrix operand order checked by reading, not run on hardware
//
#include <hip/hip_runtime.h>

// ---------------------------------------------------------------------------
// MPQBertOutput: h = hidden @ (sign(W)*alpha)^T + b ; x = h + resid ; LN(x)
// B=16, S=512 -> M=8192 rows; K=D_IN=4096; N=D_OUT=1024
// ---------------------------------------------------------------------------

#define DIN   4096
#define DOUT  1024
#define MROWS 8192   // 16*512

typedef __bf16 v16bf __attribute__((ext_vector_type(16)));
typedef __bf16 v8bf  __attribute__((ext_vector_type(8)));
typedef float  v8f   __attribute__((ext_vector_type(8)));

// CDNA5 async global->LDS copy (ASYNCcnt) if this toolchain exposes it.
#if __has_builtin(__builtin_amdgcn_global_load_async_to_lds_b128)
#define USE_ASYNC_LDS 1
// Builtin signature: (v4i addrspace(1)*, v4i addrspace(3)*, imm int, imm int)
typedef int v4i_vs __attribute__((vector_size(16)));
typedef __attribute__((address_space(1))) v4i_vs* async_gptr;
typedef __attribute__((address_space(3))) v4i_vs* async_lptr;
#endif

__device__ __forceinline__ void wait_async0() {
#if __has_builtin(__builtin_amdgcn_s_wait_asynccnt)
    __builtin_amdgcn_s_wait_asynccnt(0);
#else
    asm volatile("s_wait_asynccnt 0x0" ::: "memory");
#endif
}

// fp32 -> bf16 bits, round-to-nearest-even
__device__ __forceinline__ unsigned short f32_to_bf16_bits(float f) {
    unsigned u = __float_as_uint(f);
    unsigned r = u + 0x7FFFu + ((u >> 16) & 1u);
    return (unsigned short)(r >> 16);
}

// ---------------------------------------------------------------------------
// Kernel 1: per-output-row weight binarization: Wq[o,k] = bf16(sign(W)*alpha)
// ---------------------------------------------------------------------------
__global__ __launch_bounds__(256)
void quantize_w_kernel(const float* __restrict__ W,
                       unsigned short* __restrict__ Wq) {
    __shared__ float red[256];
    const int o   = blockIdx.x;
    const int tid = threadIdx.x;
    const float* wrow = W + (size_t)o * DIN;

    float s = 0.f;
    for (int k = tid; k < DIN; k += 256) s += fabsf(wrow[k]);
    red[tid] = s;
    __syncthreads();
    for (int off = 128; off > 0; off >>= 1) {
        if (tid < off) red[tid] += red[tid + off];
        __syncthreads();
    }
    const float alpha = red[0] * (1.0f / (float)DIN);
    const unsigned short abits = f32_to_bf16_bits(alpha);  // alpha >= 0

    unsigned short* qrow = Wq + (size_t)o * DIN;
    for (int k = tid; k < DIN; k += 256) {
        const unsigned sign = __float_as_uint(wrow[k]) & 0x80000000u;
        qrow[k] = (unsigned short)(abits | (sign >> 16));
    }
}

// ---------------------------------------------------------------------------
// Kernel 2: hidden_states fp32 -> bf16 (vectorized: float4 in, 8B out)
// ---------------------------------------------------------------------------
__global__ __launch_bounds__(256)
void cvt_a_kernel(const float* __restrict__ A,
                  unsigned short* __restrict__ Ab) {
    const size_t i = ((size_t)blockIdx.x * 256 + threadIdx.x) * 4;
    const float4 v = *reinterpret_cast<const float4*>(A + i);
    uint2 p;
    p.x = (unsigned)f32_to_bf16_bits(v.x) | ((unsigned)f32_to_bf16_bits(v.y) << 16);
    p.y = (unsigned)f32_to_bf16_bits(v.z) | ((unsigned)f32_to_bf16_bits(v.w) << 16);
    *reinterpret_cast<uint2*>(Ab + i) = p;
}

// ---------------------------------------------------------------------------
// Kernel 3: bf16 WMMA GEMM, 128x128 tile per workgroup, fused +bias +residual
//   8 waves (wave32) in 2(M) x 4(N); each wave: 4x2 fragments of 16x16.
//   K-depth 64 per stage (16 WMMAs per barrier), double-buffered LDS filled
//   with GLOBAL_LOAD_ASYNC_TO_LDS_B128 (ASYNCcnt). Stage selected by XOR'd
//   LDS offset; k-loop kept at unroll 1 so accumulators stay pinned in place.
// ---------------------------------------------------------------------------
#define KDEPTH   64                  // bf16 K elements per stage
#define LDSPITCH 72                  // ushorts per row: 64 data + 8 pad (144B)
#define STAGEOFF (128 * LDSPITCH)    // ushorts per stage per matrix

__global__ __launch_bounds__(256)
void gemm_wmma_kernel(const unsigned short* __restrict__ Abf,
                      const unsigned short* __restrict__ Wq,
                      const float* __restrict__ bias,
                      const float* __restrict__ resid,
                      float* __restrict__ out) {
    __shared__ unsigned short ldsA[2 * STAGEOFF];
    __shared__ unsigned short ldsB[2 * STAGEOFF];

    const int tid   = threadIdx.x;
    const int lane  = tid & 31;
    const int wave  = tid >> 5;            // 0..7
    const int waveM = (wave & 1) * 64;     // 0, 64
    const int waveN = (wave >> 1) * 32;    // 0, 32, 64, 96

    const int nBase = blockIdx.x * 128;
    const int mBase = blockIdx.y * 128;

    const v8f vzero = {0.f, 0.f, 0.f, 0.f, 0.f, 0.f, 0.f, 0.f};
    v8f acc[4][2];
    #pragma unroll
    for (int mi = 0; mi < 4; ++mi)
        #pragma unroll
        for (int ni = 0; ni < 2; ++ni) acc[mi][ni] = vzero;

    // Tile fill: 128 rows x 64 bf16 per matrix = 1024 x 16B chunks; 4/thread.
#if defined(USE_ASYNC_LDS)
    auto issue_stage = [&](int kt, unsigned stBase) {
        const int kBase = kt * KDEPTH;
        #pragma unroll
        for (int i = 0; i < 4; ++i) {
            const int c   = tid + i * 256;   // 0..1023
            const int row = c >> 3;
            const int cc  = c & 7;
            const unsigned short* ga = Abf + (size_t)(mBase + row) * DIN + kBase + cc * 8;
            const unsigned short* gb = Wq  + (size_t)(nBase + row) * DIN + kBase + cc * 8;
            const unsigned loff = stBase + row * LDSPITCH + cc * 8;
            __builtin_amdgcn_global_load_async_to_lds_b128(
                (async_gptr)ga, (async_lptr)&ldsA[loff], 0, 0);
            __builtin_amdgcn_global_load_async_to_lds_b128(
                (async_gptr)gb, (async_lptr)&ldsB[loff], 0, 0);
        }
    };
#else
    uint4 aReg[4], bReg[4];
    auto gload = [&](int kt) {
        const int kBase = kt * KDEPTH;
        #pragma unroll
        for (int i = 0; i < 4; ++i) {
            const int c   = tid + i * 256;
            const int row = c >> 3;
            const int cc  = c & 7;
            aReg[i] = *reinterpret_cast<const uint4*>(
                Abf + (size_t)(mBase + row) * DIN + kBase + cc * 8);
            bReg[i] = *reinterpret_cast<const uint4*>(
                Wq + (size_t)(nBase + row) * DIN + kBase + cc * 8);
        }
    };
    auto lstore = [&](unsigned stBase) {
        #pragma unroll
        for (int i = 0; i < 4; ++i) {
            const int c   = tid + i * 256;
            const int row = c >> 3;
            const int cc  = c & 7;
            const unsigned loff = stBase + row * LDSPITCH + cc * 8;
            *reinterpret_cast<uint4*>(&ldsA[loff]) = aReg[i];
            *reinterpret_cast<uint4*>(&ldsB[loff]) = bReg[i];
        }
    };
#endif

    const int fr = lane & 15;
    const int kh = (lane >> 4) * 8;  // low-half K offset (bf16 elements)

    // Compute one stage: 2 x (2 resident B frags + streamed A frags), 16 WMMAs
    auto compute = [&](unsigned stBase) {
        #pragma unroll
        for (int kk = 0; kk < 2; ++kk) {
            const unsigned ko = stBase + kk * 32 + kh;
            v16bf bfrag[2];
            #pragma unroll
            for (int ni = 0; ni < 2; ++ni) {
                const unsigned short* p = &ldsB[ko + (waveN + ni * 16 + fr) * LDSPITCH];
                v8bf lo = *reinterpret_cast<const v8bf*>(p);
                v8bf hi = *reinterpret_cast<const v8bf*>(p + 16);
                bfrag[ni] = __builtin_shufflevector(lo, hi,
                    0,1,2,3,4,5,6,7,8,9,10,11,12,13,14,15);
            }
            #pragma unroll
            for (int mi = 0; mi < 4; ++mi) {
                const unsigned short* p = &ldsA[ko + (waveM + mi * 16 + fr) * LDSPITCH];
                v8bf lo = *reinterpret_cast<const v8bf*>(p);
                v8bf hi = *reinterpret_cast<const v8bf*>(p + 16);
                v16bf a = __builtin_shufflevector(lo, hi,
                    0,1,2,3,4,5,6,7,8,9,10,11,12,13,14,15);
                acc[mi][0] = __builtin_amdgcn_wmma_f32_16x16x32_bf16(
                    false, a, false, bfrag[0], (short)0, acc[mi][0], false, false);
                acc[mi][1] = __builtin_amdgcn_wmma_f32_16x16x32_bf16(
                    false, a, false, bfrag[1], (short)0, acc[mi][1], false, false);
            }
        }
    };

    const int KT = DIN / KDEPTH;  // 64 stages
    unsigned off = 0;

#if defined(USE_ASYNC_LDS)
    issue_stage(0, 0);
    wait_async0();
    __syncthreads();
    #pragma unroll 1
    for (int kt = 0; kt < KT - 1; ++kt) {
        issue_stage(kt + 1, off ^ STAGEOFF);
        compute(off);
        wait_async0();
        __syncthreads();
        off ^= STAGEOFF;
    }
    compute(off);
#else
    gload(0);
    lstore(0);
    __syncthreads();
    #pragma unroll 1
    for (int kt = 0; kt < KT - 1; ++kt) {
        gload(kt + 1);
        compute(off);
        lstore(off ^ STAGEOFF);
        __syncthreads();
        off ^= STAGEOFF;
    }
    compute(off);
#endif

    // Epilogue: x = acc + bias[n] + resid[m,n] -> out (LayerNorm kernel follows)
    // C/D layout: lanes 0-15: (M=r, N=lane); lanes 16-31: (M=8+r, N=lane-16)
    const int nLane    = lane & 15;
    const int mOffLane = (lane >> 4) * 8;
    #pragma unroll
    for (int mi = 0; mi < 4; ++mi) {
        #pragma unroll
        for (int ni = 0; ni < 2; ++ni) {
            #pragma unroll
            for (int r = 0; r < 8; ++r) {
                const int m = mBase + waveM + mi * 16 + mOffLane + r;
                const int n = nBase + waveN + ni * 16 + nLane;
                const size_t idx = (size_t)m * DOUT + n;
                out[idx] = acc[mi][ni][r] + bias[n] + resid[idx];
            }
        }
    }
}

// ---------------------------------------------------------------------------
// Kernel 4: in-place LayerNorm over last dim (1024) — one block per row
// ---------------------------------------------------------------------------
__global__ __launch_bounds__(256)
void layernorm_kernel(float* __restrict__ out,
                      const float* __restrict__ gamma,
                      const float* __restrict__ beta) {
    __shared__ float s1[256];
    __shared__ float s2[256];
    const int row = blockIdx.x;
    const int tid = threadIdx.x;
    float* rp = out + (size_t)row * DOUT;

    const float4 v = *reinterpret_cast<const float4*>(rp + tid * 4);
    s1[tid] = v.x + v.y + v.z + v.w;
    s2[tid] = v.x * v.x + v.y * v.y + v.z * v.z + v.w * v.w;
    __syncthreads();
    for (int off = 128; off > 0; off >>= 1) {
        if (tid < off) { s1[tid] += s1[tid + off]; s2[tid] += s2[tid + off]; }
        __syncthreads();
    }
    const float mu  = s1[0] * (1.0f / (float)DOUT);
    const float var = s2[0] * (1.0f / (float)DOUT) - mu * mu;
    const float inv = rsqrtf(var + 1e-12f);

    const float4 g = *reinterpret_cast<const float4*>(gamma + tid * 4);
    const float4 b = *reinterpret_cast<const float4*>(beta + tid * 4);
    float4 o;
    o.x = g.x * (v.x - mu) * inv + b.x;
    o.y = g.y * (v.y - mu) * inv + b.y;
    o.z = g.z * (v.z - mu) * inv + b.z;
    o.w = g.w * (v.w - mu) * inv + b.w;
    *reinterpret_cast<float4*>(rp + tid * 4) = o;
}

// ---------------------------------------------------------------------------
extern "C" void kernel_launch(void* const* d_in, const int* in_sizes, int n_in,
                              void* d_out, int out_size, void* d_ws, size_t ws_size,
                              hipStream_t stream) {
    const float* hidden = (const float*)d_in[0];  // [16,512,4096]
    const float* resid  = (const float*)d_in[1];  // [16,512,1024]
    const float* W      = (const float*)d_in[2];  // [1024,4096]
    const float* bias   = (const float*)d_in[3];  // [1024]
    const float* gamma  = (const float*)d_in[4];  // [1024]
    const float* beta   = (const float*)d_in[5];  // [1024]
    float* out = (float*)d_out;                   // [16,512,1024]

    // Workspace: bf16 A (64 MB) + bf16 quantized W (8 MB)
    unsigned short* Abf = (unsigned short*)d_ws;
    unsigned short* Wq  = Abf + (size_t)MROWS * DIN;

    quantize_w_kernel<<<DOUT, 256, 0, stream>>>(W, Wq);

    const int cvt_blocks = (int)(((size_t)MROWS * DIN) / (4 * 256));  // 32768
    cvt_a_kernel<<<cvt_blocks, 256, 0, stream>>>(hidden, Abf);

    dim3 grid(DOUT / 128, MROWS / 128);  // 8 x 64
    gemm_wmma_kernel<<<grid, 256, 0, stream>>>(Abf, Wq, bias, resid, out);

    layernorm_kernel<<<MROWS, 256, 0, stream>>>(out, gamma, beta);
}
